// GRUModel_73959336837602
// MI455X (gfx1250) — matrix operand order repaired and checked by reading, not compiled, not run on hardware
//
#include <hip/hip_runtime.h>

typedef __attribute__((ext_vector_type(16))) _Float16     v16h;
typedef __attribute__((ext_vector_type(8)))  _Float16     v8h;
typedef __attribute__((ext_vector_type(8)))  float        v8f;
typedef __attribute__((ext_vector_type(4)))  int          v4i;
typedef __attribute__((ext_vector_type(4)))  unsigned int v4u;
typedef __attribute__((ext_vector_type(8)))  unsigned int v8u;

#define AS1 __attribute__((address_space(1)))
#define AS3 __attribute__((address_space(3)))

#define Bsz 65536
#define Tt  30
#define In  26
#define Hh  24
#define Fc1 8
#define Ncls 4
#define XSLAB (Tt * 16 * In)   // 12480 fp32 elements per block

#ifndef USE_TDM
#define USE_TDM 1              // flip to 0 if the tensor_load_to_lds asm is rejected
#endif

#if __has_builtin(__builtin_amdgcn_global_load_async_to_lds_b128) && \
    __has_builtin(__builtin_amdgcn_s_wait_asynccnt)
#define USE_ASYNC_LDS 1
#else
#define USE_ASYNC_LDS 0
#endif

// ---------- fast device math (lowers to v_exp_f32 / v_rcp_f32 TRANS ops) ----------
__device__ __forceinline__ float fast_sigmoid(float x) {
    return __builtin_amdgcn_rcpf(1.0f + __builtin_amdgcn_exp2f(-1.44269504f * x));
}
__device__ __forceinline__ float fast_tanh(float x) {
    x = fminf(10.0f, fmaxf(-10.0f, x));
    float e = __builtin_amdgcn_exp2f(2.88539008f * x);   // exp(2x)
    return (e - 1.0f) * __builtin_amdgcn_rcpf(e + 1.0f);
}

// Build one 16x32 f16 A-fragment (M = output rows, K = reduction dim).
// Bias folded in as column k == ncols (pairs with the constant 1.0 planted in
// the B-operand vectors). Rows >= nrows and cols > ncols are zero.
__device__ __forceinline__ v16h make_a_tile(const float* __restrict__ W,
                                            const float* __restrict__ bias,
                                            int g0, int local, int nrows, int ncols,
                                            int kb1) {
    v16h a;
#pragma unroll
    for (int e = 0; e < 16; ++e) {
        int k = 2 * kb1 + e;
        float v = 0.0f;
        if (local < nrows) {
            if (k < ncols)        v = W[(g0 + local) * ncols + k];
            else if (k == ncols)  v = bias[g0 + local];
        }
        a[e] = (_Float16)v;
    }
    return a;
}

__device__ __forceinline__ v8f wmma16(v16h a, v16h b, v8f c) {
    return __builtin_amdgcn_wmma_f32_16x16x32_f16(false, a, false, b, (short)0, c,
                                                  false, false);
}

__global__ __launch_bounds__(32)
void gru_fused_kernel(const float* __restrict__ x,
                      const float* __restrict__ W_ih,
                      const float* __restrict__ W_hh,
                      const float* __restrict__ b_ih,
                      const float* __restrict__ b_hh,
                      const float* __restrict__ fc1_w,
                      const float* __restrict__ fc1_b,
                      const float* __restrict__ fc2_w,
                      const float* __restrict__ fc2_b,
                      float* __restrict__ out) {
    // raw fp32 x slab for this block: [n][t][i] contiguous (+pad for OOB-safe b64 reads)
    __attribute__((aligned(16))) __shared__ float sXraw[XSLAB + 16];
    // hidden state shuttle: [n][32] f16 with k==24 -> 1.0 (b_hh / fc1_b K-fold)
    __attribute__((aligned(16))) __shared__ _Float16 sH[16 * 32];

    const int lane = threadIdx.x;      // wave32
    const int row  = lane & 15;        // batch lane (N) / M row for A tiles
    const int hi   = lane >> 4;
    const int kb1  = hi * 8;           // half the A fragment K base (k = 2*kb1 + e)
    const int koff = hi * 16;          // B fragment K base
    const int b0   = blockIdx.x * 16;  // 16 batch rows per wave

    const float* xg = x + (size_t)b0 * (Tt * In);

    // ---------------- stream x slab into LDS ----------------
#if USE_TDM
    {
        // Tensor DMA Descriptor: 1D tensor, data_size=4B, dim0=tile0=12480,
        // count=1, type=2, no multicast / iterate / pad / atomic-barrier.
        unsigned long long ga = (unsigned long long)(uintptr_t)xg;
        unsigned int ldsoff   = (unsigned int)(uintptr_t)(void*)sXraw;
        v4u g0;
        g0[0] = 1u;                                            // count=1, user mode
        g0[1] = ldsoff;                                        // lds_addr
        g0[2] = (unsigned int)ga;                              // global_addr[31:0]
        g0[3] = ((unsigned int)(ga >> 32) & 0x01FFFFFFu) |     // global_addr[56:32]
                (2u << 30);                                    // type=2 ("image")
        v8u g1;
        g1[0] = 0x00020000u;                                   // data_size=2 (4B)
        g1[1] = (unsigned int)(XSLAB & 0xFFFFu) << 16;         // tensor_dim0 lo
        g1[2] = ((unsigned int)(XSLAB >> 16) & 0xFFFFu) |      // tensor_dim0 hi
                (1u << 16);                                    // tensor_dim1 = 1
        g1[3] = (unsigned int)XSLAB << 16;                     // tile_dim0 = 12480
        g1[4] = 1u;                                            // tile_dim1=1, tile_dim2=0
        g1[5] = (unsigned int)XSLAB;                           // dim0_stride lo
        g1[6] = (unsigned int)(XSLAB & 0xFFFFu) << 16;         // dim1_stride lo
        g1[7] = 0u;
        asm volatile("tensor_load_to_lds %0, %1" :: "s"(g0), "s"(g1) : "memory");
        __builtin_amdgcn_s_wait_tensorcnt(0);
    }
#elif USE_ASYNC_LDS
    {
        const char* gsrc = (const char*)xg;            // 16B aligned: b0*3120
        char* ldst = (char*)sXraw;
        for (int i = lane; i < XSLAB / 4; i += 32) {
            __builtin_amdgcn_global_load_async_to_lds_b128(
                (AS1 v4i*)(gsrc + (size_t)i * 16),
                (AS3 v4i*)(ldst + i * 16), 0, 0);
        }
        __builtin_amdgcn_s_wait_asynccnt(0);
    }
#else
    for (int i = lane; i < XSLAB; i += 32) sXraw[i] = xg[i];
#endif

    // ---------------- resident weight fragments (loop-invariant) ----------------
    // tiles 0,1 = r rows(0..23 pad 32) ; 2,3 = z ; 4,5 = n
    v16h aWih[6], aWhh[6];
#pragma unroll
    for (int tgt = 0; tgt < 6; ++tgt) {
        int gate  = tgt >> 1;
        int local = (tgt & 1) * 16 + row;
        aWih[tgt] = make_a_tile(W_ih, b_ih, gate * Hh, local, Hh, In, kb1);
        aWhh[tgt] = make_a_tile(W_hh, b_hh, gate * Hh, local, Hh, Hh, kb1);
    }
    v16h aF1 = make_a_tile(fc1_w, fc1_b, 0, row, Fc1, Hh, kb1);
    v16h aF2 = make_a_tile(fc2_w, fc2_b, 0, row, Ncls, Fc1, kb1);

    // init h = 0 with 1.0 at k==Hh (bias-fold one)
    for (int idx = lane; idx < 16 * 32; idx += 32) {
        int k = idx & 31;
        sH[idx] = (k == Hh) ? (_Float16)1.0f : (_Float16)0.0f;
    }
    __syncthreads();

    v8f Hreg[2];
#pragma unroll
    for (int p = 0; p < 2; ++p)
#pragma unroll
        for (int v = 0; v < 8; ++v) Hreg[p][v] = 0.0f;

    v8f zeroC{};

    // ---------------- recurrence: 12 WMMAs / step ----------------
#pragma unroll 1
    for (int t = 0; t < Tt; ++t) {
        // assemble x B-operand from raw fp32 slab: lane holds K = koff..koff+15
        float xv[16];
        {
            int base = row * (Tt * In) + t * In + koff;   // 8B aligned (even dwords)
#pragma unroll
            for (int j = 0; j < 8; ++j) {
                float2 t2 = *(const float2*)&sXraw[base + 2 * j];
                xv[2 * j]     = t2.x;
                xv[2 * j + 1] = t2.y;
            }
        }
        v16h bx;
#pragma unroll
        for (int e = 0; e < 16; ++e) {
            float v = xv[e];
            if (hi) {                       // k = 16+e : 16..25 data, 26 -> 1, >26 -> 0
                v = (e < 10) ? xv[e] : ((e == 10) ? 1.0f : 0.0f);
            }
            bx[e] = (_Float16)v;
        }
        v16h bh = *(const v16h*)&sH[row * 32 + koff];

        v8f accRZ[4], accXN[2], accHN[2];
#pragma unroll
        for (int i = 0; i < 4; ++i) {              // r,z: x-proj then h-proj chained
            accRZ[i] = wmma16(aWih[i], bx, zeroC);
            accRZ[i] = wmma16(aWhh[i], bh, accRZ[i]);
        }
#pragma unroll
        for (int p = 0; p < 2; ++p) {              // n gate needs xn / hn separate
            accXN[p] = wmma16(aWih[4 + p], bx, zeroC);
            accHN[p] = wmma16(aWhh[4 + p], bh, zeroC);
        }

#pragma unroll
        for (int p = 0; p < 2; ++p) {
            v8h hpack;
#pragma unroll
            for (int v = 0; v < 8; ++v) {
                float r = fast_sigmoid(accRZ[p][v]);
                float z = fast_sigmoid(accRZ[2 + p][v]);
                float n = fast_tanh(accXN[p][v] + r * accHN[p][v]);
                float hn = (1.0f - z) * n + z * Hreg[p][v];
                Hreg[p][v] = hn;
                hpack[v] = (_Float16)hn;
            }
            // D-layout -> B-layout shuttle: lane holds rows j = 16p + v + 8*hi
            if (p == 0) {
                *(v8h*)&sH[row * 32 + hi * 8] = hpack;   // j = 0..15
            } else if (hi == 0) {
                *(v8h*)&sH[row * 32 + 16] = hpack;       // j = 16..23 (24..31 keep 1/0)
            }
        }
    }

    // ---------------- FC head: 2 WMMAs + lane-local softmax ----------------
    v16h bhf = *(const v16h*)&sH[row * 32 + koff];
    v8f hid  = wmma16(aF1, bhf, zeroC);
    if (hi == 0) {                                  // valid fc1 rows m = 0..7
        v8h hp;
#pragma unroll
        for (int v = 0; v < 8; ++v) hp[v] = (_Float16)fmaxf(hid[v], 0.0f);
        *(v8h*)&sH[row * 32 + 0] = hp;              // hid k=0..7
        v8h one{};
        one[0] = (_Float16)1.0f;                    // k==8 -> 1.0 (fc2_b fold)
        *(v8h*)&sH[row * 32 + 8] = one;             // k=8..15
    }
    v16h bhid = *(const v16h*)&sH[row * 32 + koff];
    v8f logit = wmma16(aF2, bhid, zeroC);

    if (hi == 0) {                                  // classes live in vgpr 0..3, lanes 0..15
        float l0 = logit[0], l1 = logit[1], l2 = logit[2], l3 = logit[3];
        float m  = fmaxf(fmaxf(l0, l1), fmaxf(l2, l3));
        float e0 = __builtin_amdgcn_exp2f((l0 - m) * 1.44269504f);
        float e1 = __builtin_amdgcn_exp2f((l1 - m) * 1.44269504f);
        float e2 = __builtin_amdgcn_exp2f((l2 - m) * 1.44269504f);
        float e3 = __builtin_amdgcn_exp2f((l3 - m) * 1.44269504f);
        float rs = __builtin_amdgcn_rcpf(e0 + e1 + e2 + e3);
        float4 o = make_float4(e0 * rs, e1 * rs, e2 * rs, e3 * rs);
        *(float4*)&out[(size_t)(b0 + row) * Ncls] = o;
    }
}

extern "C" void kernel_launch(void* const* d_in, const int* in_sizes, int n_in,
                              void* d_out, int out_size, void* d_ws, size_t ws_size,
                              hipStream_t stream) {
    (void)in_sizes; (void)n_in; (void)out_size; (void)d_ws; (void)ws_size;
    const float* x     = (const float*)d_in[0];
    const float* W_ih  = (const float*)d_in[1];
    const float* W_hh  = (const float*)d_in[2];
    const float* b_ih  = (const float*)d_in[3];
    const float* b_hh  = (const float*)d_in[4];
    const float* fc1_w = (const float*)d_in[5];
    const float* fc1_b = (const float*)d_in[6];
    const float* fc2_w = (const float*)d_in[7];
    const float* fc2_b = (const float*)d_in[8];
    float* out = (float*)d_out;

    dim3 grid(Bsz / 16);
    dim3 block(32);
    gru_fused_kernel<<<grid, block, 0, stream>>>(x, W_ih, W_hh, b_ih, b_hh,
                                                 fc1_w, fc1_b, fc2_w, fc2_b, out);
}
